// UnfusedGemma4TextExperts_29678224016102
// MI455X (gfx1250) — matrix operand order repaired and checked by reading, not compiled, not run on hardware
//
#include <hip/hip_runtime.h>

// ---------------------------------------------------------------------------
// Fused MoE (8 experts, uniform 1/8 routing):
//   for e: OUT += ( gelu_tanh(X @ gw[e]^T) * (X @ uw[e]^T) ) @ dw[e]^T / 8
// T=8192, H=2048, F=1024.  v_wmma_f32_16x16x32_f16, fp32 accumulate.
//
// Fast path (ws >= 128 MiB): one-time f32->f16 conversion of X and all
// weights into ws, so the GEMM hot loop is pure b128 loads + WMMA.
// Fallback: convert f32->f16 inline (v_cvt_pk_f16_f32) per fragment.
// ---------------------------------------------------------------------------

typedef __attribute__((ext_vector_type(16))) _Float16 v16h;
typedef __attribute__((ext_vector_type(8)))  _Float16 v8h;
typedef __attribute__((ext_vector_type(4)))  _Float16 v4h;
typedef __attribute__((ext_vector_type(8)))  float    v8f;
typedef __attribute__((ext_vector_type(4)))  float    v4f;

#define TK 8192   // tokens
#define HD 2048   // hidden
#define FD 1024   // intermediate
#define NE 8      // experts
#define BT 64     // token rows per workgroup
#define BK 128    // K-chunk staged in LDS

#define XS_STR  136    // 128 + 8 f16 pad  (staged X tile stride)
#define ACT_STR 1032   // 1024 + 8 f16 pad (activation tile stride)

__device__ __forceinline__ v8f wmma_f16(v16h a, v16h b, v8f c) {
  return __builtin_amdgcn_wmma_f32_16x16x32_f16(false, a, false, b,
                                                (short)0, c, false, false);
}

// B-fragment: 16 consecutive f32 -> f16 (fallback path).
__device__ __forceinline__ v16h cvt_row16(const float* __restrict__ p) {
  v4f f0 = ((const v4f*)p)[0];
  v4f f1 = ((const v4f*)p)[1];
  v4f f2 = ((const v4f*)p)[2];
  v4f f3 = ((const v4f*)p)[3];
  v16h r;
  r[0]=(_Float16)f0[0];  r[1]=(_Float16)f0[1];  r[2]=(_Float16)f0[2];  r[3]=(_Float16)f0[3];
  r[4]=(_Float16)f1[0];  r[5]=(_Float16)f1[1];  r[6]=(_Float16)f1[2];  r[7]=(_Float16)f1[3];
  r[8]=(_Float16)f2[0];  r[9]=(_Float16)f2[1];  r[10]=(_Float16)f2[2]; r[11]=(_Float16)f2[3];
  r[12]=(_Float16)f3[0]; r[13]=(_Float16)f3[1]; r[14]=(_Float16)f3[2]; r[15]=(_Float16)f3[3];
  return r;
}

// B-fragment: 16 consecutive f16 (fast path) -> two b128 loads, no VALU.
__device__ __forceinline__ v16h load_row16_f16(const _Float16* __restrict__ p) {
  v16h r;
  ((v8h*)&r)[0] = *(const v8h*)(p);
  ((v8h*)&r)[1] = *(const v8h*)(p + 8);
  return r;
}

// A-fragment (16x32 f16) from LDS row-major f16 storage.
// lane<16: row M=lane, K = k..k+15
// lane>=16: row M=lane-16, K = k+16..k+31
__device__ __forceinline__ v16h lds_a_frag(const _Float16* base) {
  v16h a;
  ((v8h*)&a)[0] = *(const v8h*)(base);
  ((v8h*)&a)[1] = *(const v8h*)(base + 8);
  return a;
}

__device__ __forceinline__ float gelu_tanh(float x) {
  float x3 = x * x * x;
  float t  = tanhf(0.7978845608028654f * (x + 0.044715f * x3));
  return 0.5f * x * (1.0f + t);
}

// f32 -> f16 bulk conversion (prep pass). n must be a multiple of 8.
__global__ void __launch_bounds__(256)
cvt_f32_f16_kernel(const float* __restrict__ src, _Float16* __restrict__ dst, int n)
{
  int i = (blockIdx.x * 256 + threadIdx.x) * 8;
  if (i >= n) return;
  v4f a = ((const v4f*)(src + i))[0];
  v4f b = ((const v4f*)(src + i))[1];
  v8h h;
  h[0]=(_Float16)a[0]; h[1]=(_Float16)a[1]; h[2]=(_Float16)a[2]; h[3]=(_Float16)a[3];
  h[4]=(_Float16)b[0]; h[5]=(_Float16)b[1]; h[6]=(_Float16)b[2]; h[7]=(_Float16)b[3];
  *(v8h*)(dst + i) = h;
}

template <bool F16W>
__global__ void __launch_bounds__(256)
moe_fused_kernel(const void* __restrict__ Xp,   // [TK, HD]      f32 or f16
                 const void* __restrict__ GWp,  // [NE, FD, HD]  f32 or f16
                 const void* __restrict__ UWp,  // [NE, FD, HD]  f32 or f16
                 const void* __restrict__ DWp,  // [NE, HD, FD]  f32 or f16
                 float* __restrict__ OUT)       // [TK, HD] (pre-zeroed)
{
  __shared__ _Float16 Xs[BT * XS_STR];     // 17 KB   staged X K-chunk (f16)
  __shared__ _Float16 ACT[BT * ACT_STR];   // 129 KB  gelu(G)*U tile (f16)

  const int tile = blockIdx.x;           // token tile (128 of them)
  const int e    = blockIdx.y;           // expert
  const int t0   = tile * BT;
  const int tid  = threadIdx.x;
  const int wave = tid >> 5;
  const int lane = tid & 31;
  const int l16  = lane & 15;
  const int hi   = lane >> 4;            // 0: lanes 0-15, 1: lanes 16-31

  const size_t eFH = (size_t)e * FD * HD;

  const int r  = wave & 3;               // row tile 0..3 (16 rows each)
  const int cg = wave >> 2;              // column-half owner (0/1)

  // ------------------- Phase 1: ACT = gelu(X@gw^T) * (X@uw^T) ---------------
  for (int fs = 0; fs < FD; fs += 128) {        // F-strip: 8 col-tiles, split by cg
    const int c0 = fs + cg * 64;                // this wave: 4 col-tiles
    v8f accG[4] = {}; v8f accU[4] = {};

    for (int k0 = 0; k0 < HD; k0 += BK) {
      __syncthreads();                          // Xs consumers from prev chunk done
      if constexpr (F16W) {                     // stage 64x128 f16: pure copy
        const _Float16* Xh = (const _Float16*)Xp;
        const int row = tid >> 2;
        const _Float16* srow = Xh + (size_t)(t0 + row) * HD + k0;
        #pragma unroll
        for (int j = 0; j < 4; ++j) {
          const int c8 = (tid & 3) + j * 4;     // v8h chunk index (16 per row)
          *(v8h*)&Xs[row * XS_STR + c8 * 8] = ((const v8h*)srow)[c8];
        }
      } else {                                  // stage 64x128 f32 -> f16
        const float* Xf = (const float*)Xp;
        const int row = tid >> 2;
        const float* srow = Xf + (size_t)(t0 + row) * HD + k0;
        #pragma unroll
        for (int j = 0; j < 8; ++j) {
          const int c4 = (tid & 3) + j * 4;     // float4 chunk index (32 per row)
          v4f v = ((const v4f*)srow)[c4];
          v4h h;
          h[0]=(_Float16)v[0]; h[1]=(_Float16)v[1];
          h[2]=(_Float16)v[2]; h[3]=(_Float16)v[3];
          *(v4h*)&Xs[row * XS_STR + c4 * 4] = h;
        }
      }
      __syncthreads();

      #pragma unroll
      for (int kk = 0; kk < BK; kk += 32) {
        const v16h a = lds_a_frag(&Xs[(r * 16 + l16) * XS_STR + kk + hi * 16]);
        #pragma unroll
        for (int c = 0; c < 4; ++c) {
          const int n = c0 + c * 16 + l16;      // weight row = output feature
          const size_t off = eFH + (size_t)n * HD + (size_t)(k0 + kk + hi * 16);
          v16h bg, bu;
          if constexpr (F16W) {
            bg = load_row16_f16((const _Float16*)GWp + off);
            bu = load_row16_f16((const _Float16*)UWp + off);
          } else {
            bg = cvt_row16((const float*)GWp + off);
            bu = cvt_row16((const float*)UWp + off);
          }
          accG[c] = wmma_f16(a, bg, accG[c]);
          accU[c] = wmma_f16(a, bu, accU[c]);
        }
      }
    }

    // epilogue: ACT[m][n] = gelu(G)*U  (C layout: lane<16 -> M=v, lane>=16 -> M=v+8)
    #pragma unroll
    for (int c = 0; c < 4; ++c) {
      const int n = c0 + c * 16 + l16;
      #pragma unroll
      for (int v = 0; v < 8; ++v) {
        const int m = r * 16 + v + hi * 8;
        const float act = gelu_tanh(accG[c][v]) * accU[c][v];
        ACT[m * ACT_STR + n] = (_Float16)act;
      }
    }
  }
  __syncthreads();   // ACT fully written before cross-wave reads

  // ------------------- Phase 2: OUT += ACT @ dw^T * (1/8) -------------------
  const size_t eHF = (size_t)e * HD * FD;
  for (int cs = 0; cs < 1024; cs += 64) {       // wave covers cols cg*1024 + cs ..
    const int c0 = cg * 1024 + cs;
    v8f acc[4] = {};
    for (int k = 0; k < FD; k += 32) {
      const v16h a = lds_a_frag(&ACT[(r * 16 + l16) * ACT_STR + k + hi * 16]);
      #pragma unroll
      for (int c = 0; c < 4; ++c) {
        const int n = c0 + c * 16 + l16;        // dw row = output hidden index
        const size_t off = eHF + (size_t)n * FD + (size_t)(k + hi * 16);
        v16h b;
        if constexpr (F16W) b = load_row16_f16((const _Float16*)DWp + off);
        else                b = cvt_row16((const float*)DWp + off);
        acc[c] = wmma_f16(a, b, acc[c]);
      }
    }
    #pragma unroll
    for (int c = 0; c < 4; ++c) {
      const int n = c0 + c * 16 + l16;
      #pragma unroll
      for (int v = 0; v < 8; ++v) {
        const int m = r * 16 + v + hi * 8;
        unsafeAtomicAdd(&OUT[(size_t)(t0 + m) * HD + n], acc[c][v] * 0.125f);
      }
    }
  }
}

extern "C" void kernel_launch(void* const* d_in, const int* in_sizes, int n_in,
                              void* d_out, int out_size, void* d_ws, size_t ws_size,
                              hipStream_t stream) {
  // setup_inputs order: hidden_states, routing_weights, selected_experts,
  //                     gate_w, up_w, down_w   (routing inputs unused by reference)
  const float* X  = (const float*)d_in[0];
  const float* GW = (const float*)d_in[3];
  const float* UW = (const float*)d_in[4];
  const float* DW = (const float*)d_in[5];
  float* OUT = (float*)d_out;

  hipMemsetAsync(OUT, 0, (size_t)out_size * sizeof(float), stream);

  const size_t nX = (size_t)TK * HD;          // 16,777,216
  const size_t nW = (size_t)NE * FD * HD;     // 16,777,216 (each of GW/UW/DW)
  const size_t need = (nX + 3 * nW) * sizeof(_Float16);   // 128 MiB

  dim3 grid(TK / BT, NE);   // 128 token tiles x 8 experts = 1024 workgroups

  if (ws_size >= need) {
    _Float16* Xh  = (_Float16*)d_ws;
    _Float16* GWh = Xh + nX;
    _Float16* UWh = GWh + nW;
    _Float16* DWh = UWh + nW;
    const int blk = 256, epb = 256 * 8;
    cvt_f32_f16_kernel<<<(int)((nX + epb - 1) / epb), blk, 0, stream>>>(X,  Xh,  (int)nX);
    cvt_f32_f16_kernel<<<(int)((nW + epb - 1) / epb), blk, 0, stream>>>(GW, GWh, (int)nW);
    cvt_f32_f16_kernel<<<(int)((nW + epb - 1) / epb), blk, 0, stream>>>(UW, UWh, (int)nW);
    cvt_f32_f16_kernel<<<(int)((nW + epb - 1) / epb), blk, 0, stream>>>(DW, DWh, (int)nW);
    moe_fused_kernel<true><<<grid, 256, 0, stream>>>(Xh, GWh, UWh, DWh, OUT);
  } else {
    moe_fused_kernel<false><<<grid, 256, 0, stream>>>(X, GW, UW, DW, OUT);
  }
}
